// MambaBlockv2_76802605187532
// MI455X (gfx1250) — matrix operand order repaired and checked by reading, not compiled, not run on hardware
//
#include <hip/hip_runtime.h>
#include <hip/hip_bf16.h>

// ---------------- config (mirrors reference) ----------------
constexpr int B_SZ    = 8;
constexpr int SEQ     = 4096;
constexpr int D_MODEL = 1024;
constexpr int D_STATE = 128;
constexpr int D_CONV  = 4;
constexpr int D_INNER = 2048;
constexpr int DT_RANK = 64;
constexpr int SSM_W   = DT_RANK + 2 * D_STATE;   // 320

constexpr size_t BL   = (size_t)B_SZ * SEQ;       // 32768 rows
constexpr size_t X_N  = BL * D_MODEL;             // 33,554,432
constexpr size_t XI_N = BL * D_INNER;             // 67,108,864
constexpr size_t SS_N = BL * SSM_W;               // 10,485,760
constexpr size_t ST_N = BL * D_STATE;             // 4,194,304

constexpr size_t WIN_N  = (size_t)D_INNER * D_MODEL;  // 2,097,152
constexpr size_t WX_N   = (size_t)SSM_W * D_INNER;    //   655,360
constexpr size_t WOUT_N = (size_t)D_MODEL * D_INNER;  // 2,097,152

// ---------------- WMMA types ----------------
typedef __attribute__((ext_vector_type(16))) __bf16 v16bf;
typedef __attribute__((ext_vector_type(8)))  __bf16 v8bf;
typedef __attribute__((ext_vector_type(4)))  __bf16 v4bf;
typedef __attribute__((ext_vector_type(8)))  float  v8f;
typedef __attribute__((ext_vector_type(4)))  float  v4f;

__device__ __forceinline__ __bf16 f2bf(float f) {
  unsigned int u = __builtin_bit_cast(unsigned int, f);
  u += 0x7FFFu + ((u >> 16) & 1u);                 // round-to-nearest-even
  unsigned short s = (unsigned short)(u >> 16);
  return __builtin_bit_cast(__bf16, s);
}

// =====================================================================
// one-shot f32 -> bf16 conversion, 4 elems/thread (memory bound)
// n must be a multiple of 4 (true for all tensors here)
// =====================================================================
__global__ __launch_bounds__(256)
void cvt_bf16(const float* __restrict__ in, __bf16* __restrict__ out, size_t n4) {
  size_t i = (size_t)blockIdx.x * blockDim.x + threadIdx.x;
  if (i >= n4) return;
  v4f v = *(const v4f*)(in + i * 4);
  v4bf o;
#pragma unroll
  for (int j = 0; j < 4; ++j) o[j] = f2bf(v[j]);
  *(v4bf*)(out + i * 4) = o;
}

// =====================================================================
// GEMM: C[M,N] = A[M,K] * W[N,K]^T, A/W bf16 row-major (K contiguous),
// C f32. grid = (N/64, M/256), block = (32,4).
// Each wave: 64x64 output (4 M-tiles x 4 N-tiles, 16 WMMA accumulators).
// Per K-step: 16 global_load_b128 feed 16 v_wmma (A and B each reused 4x).
// =====================================================================
__global__ __launch_bounds__(128)
void gemm_wmma_bf16(const __bf16* __restrict__ A, const __bf16* __restrict__ W,
                    float* __restrict__ C, int N, int K) {
  const int lane   = threadIdx.x;          // 0..31
  const int wave   = threadIdx.y;          // 0..3
  const int m_base = blockIdx.y * 256 + wave * 64;
  const int n_blk  = blockIdx.x * 64;
  const int row    = lane & 15;
  const int hi     = lane >> 4;            // 0 or 1

  v8f acc[4][4] = {};

  // A fragment addressing (ISA 16x32 bf16 layout):
  // lanes 0-15 hold K {0..7,16..23}, lanes 16-31 hold K {8..15,24..31}
  const __bf16* aptr = A + (size_t)(m_base + row) * K + hi * 8;
  // B fragment addressing (32x16): col = lane&15; lanes<16 K {0..7,16..23}, lanes>=16 K {8..15,24..31}
  const __bf16* bptr = W + (size_t)(n_blk + row) * K + hi * 8;

  for (int k0 = 0; k0 < K; k0 += 32) {
    v16bf af[4];
#pragma unroll
    for (int mt = 0; mt < 4; ++mt) {
      const __bf16* ap = aptr + (size_t)(mt * 16) * K + k0;
      v8bf alo = *(const v8bf*)ap;
      v8bf ahi = *(const v8bf*)(ap + 16);
      af[mt] = __builtin_shufflevector(alo, ahi,
                 0,1,2,3,4,5,6,7,8,9,10,11,12,13,14,15);
    }
#pragma unroll
    for (int t = 0; t < 4; ++t) {
      const __bf16* bp = bptr + (size_t)(t * 16) * K + k0;
      v8bf blo = *(const v8bf*)bp;
      v8bf bhi = *(const v8bf*)(bp + 16);
      v16bf bf = __builtin_shufflevector(blo, bhi,
                   0,1,2,3,4,5,6,7,8,9,10,11,12,13,14,15);
#pragma unroll
      for (int mt = 0; mt < 4; ++mt)
        acc[mt][t] = __builtin_amdgcn_wmma_f32_16x16x32_bf16(
            false, af[mt], false, bf, (short)0, acc[mt][t], false, false);
    }
  }

  // C/D layout: VGPR r -> M = r (lanes 0-15) / r+8 (lanes 16-31), N = lane&15
#pragma unroll
  for (int mt = 0; mt < 4; ++mt) {
#pragma unroll
    for (int t = 0; t < 4; ++t) {
      int n = n_blk + t * 16 + row;
#pragma unroll
      for (int r = 0; r < 8; ++r) {
        int m = m_base + mt * 16 + hi * 8 + r;
        C[(size_t)m * N + n] = acc[mt][t][r];
      }
    }
  }
}

// =====================================================================
// Causal depthwise conv1d, K=4; emits bf16 directly (feeds GEMM2 only)
// =====================================================================
__global__ __launch_bounds__(256)
void conv_dw(const float* __restrict__ xi, const float* __restrict__ w,
             const float* __restrict__ bias, __bf16* __restrict__ xc) {
  size_t idx = (size_t)blockIdx.x * blockDim.x + threadIdx.x;
  if (idx >= XI_N) return;
  int    c  = (int)(idx % D_INNER);
  size_t bl = idx / D_INNER;
  int    l  = (int)(bl % SEQ);
  float acc = bias[c];
#pragma unroll
  for (int j = 0; j < D_CONV; ++j) {
    int ls = l - (D_CONV - 1) + j;
    if (ls >= 0)
      acc += w[c * D_CONV + j] * xi[idx + (size_t)(ls - l) * D_INNER];
  }
  xc[idx] = f2bf(acc);
}

// =====================================================================
// delta = softplus(dt @ W_dt^T + b_dt); A_bar = exp(delta*A); B_bar = delta*Bm
// =====================================================================
__global__ __launch_bounds__(256)
void ssm_prep(const float* __restrict__ ssm, const float* __restrict__ W_dt,
              const float* __restrict__ b_dt, const float* __restrict__ A_log,
              float* __restrict__ abar, float* __restrict__ bbar) {
  size_t idx = (size_t)blockIdx.x * blockDim.x + threadIdx.x;
  if (idx >= ST_N) return;
  int    n   = (int)(idx % D_STATE);
  size_t bl  = idx / D_STATE;
  const float* row = ssm + bl * SSM_W;
  float dtv = b_dt[n];
#pragma unroll 8
  for (int r = 0; r < DT_RANK; ++r) dtv += row[r] * W_dt[n * DT_RANK + r];
  float delta = (dtv > 20.f) ? dtv : log1pf(__expf(dtv));
  float Aval  = -__expf(A_log[n]);
  abar[idx] = __expf(delta * Aval);
  bbar[idx] = delta * row[DT_RANK + n];
}

// =====================================================================
// Selective scan: h_t = abar_t * h_{t-1} + bbar_t per (b,n); coalesced [b,l,n]
// =====================================================================
__global__ __launch_bounds__(128)
void scan_kernel(const float* __restrict__ abar, const float* __restrict__ bbar,
                 float* __restrict__ h) {
  int n = threadIdx.x;          // 0..127
  int b = blockIdx.x;           // 0..7
  float hv = 0.f;
  size_t base = ((size_t)b * SEQ) * D_STATE + n;
  for (int l = 0; l < SEQ; ++l) {
    size_t idx = base + (size_t)l * D_STATE;
    hv = abar[idx] * hv + bbar[idx];
    h[idx] = hv;
  }
}

// =====================================================================
// y[b,l] = w_gate * sum_n C[b,l,n] * h[b,l,n]
// =====================================================================
__global__ __launch_bounds__(256)
void y_kernel(const float* __restrict__ ssm, const float* __restrict__ h,
              const float* __restrict__ w_gate, float* __restrict__ y) {
  size_t bl = (size_t)blockIdx.x * blockDim.x + threadIdx.x;
  if (bl >= BL) return;
  const float* crow = ssm + bl * SSM_W + DT_RANK + D_STATE;
  const float* hrow = h + bl * D_STATE;
  float acc = 0.f;
#pragma unroll 8
  for (int n = 0; n < D_STATE; ++n) acc += crow[n] * hrow[n];
  y[bl] = acc * w_gate[0];
}

// =====================================================================
// g[b,l,c] = y[b,l] * silu(xi[b,l,c]); emits bf16 (feeds GEMM3 only)
// =====================================================================
__global__ __launch_bounds__(256)
void gate_kernel(const float* __restrict__ xi, const float* __restrict__ y,
                 __bf16* __restrict__ g) {
  size_t idx = (size_t)blockIdx.x * blockDim.x + threadIdx.x;
  if (idx >= XI_N) return;
  float v = xi[idx];
  float s = v / (1.f + __expf(-v));
  g[idx] = f2bf(y[idx / D_INNER] * s);
}

// =====================================================================
extern "C" void kernel_launch(void* const* d_in, const int* in_sizes, int n_in,
                              void* d_out, int out_size, void* d_ws, size_t ws_size,
                              hipStream_t stream) {
  const float* x      = (const float*)d_in[0];   // (B,L,Dm)
  const float* W_in   = (const float*)d_in[1];   // (Di,Dm)
  const float* conv_w = (const float*)d_in[2];   // (Di,1,K)
  const float* conv_b = (const float*)d_in[3];   // (Di,)
  const float* W_x    = (const float*)d_in[4];   // (320,Di)
  const float* W_dt   = (const float*)d_in[5];   // (N,64)
  const float* b_dt   = (const float*)d_in[6];   // (N,)
  const float* A_log  = (const float*)d_in[7];   // (N,)
  const float* w_gate = (const float*)d_in[8];   // (1,)
  const float* W_out  = (const float*)d_in[9];   // (Dm,Di)
  float* out = (float*)d_out;

  // ---- workspace layout ----
  float* ws_f = (float*)d_ws;
  float* xi   = ws_f;                     // XI_N f32
  float* ssm  = xi + XI_N;                // SS_N f32
  float* abar = ssm + SS_N;               // ST_N f32
  float* bbar = abar + ST_N;              // ST_N f32
  float* h    = bbar + ST_N;              // ST_N f32
  float* yv   = h + ST_N;                 // BL   f32
  __bf16* ws_b   = (__bf16*)(yv + BL);
  __bf16* x_bf   = ws_b;                  // X_N
  __bf16* xc_bf  = x_bf + X_N;            // XI_N (reused for gated act)
  __bf16* Win_bf  = xc_bf + XI_N;         // WIN_N
  __bf16* Wx_bf   = Win_bf + WIN_N;       // WX_N
  __bf16* Wout_bf = Wx_bf + WX_N;         // WOUT_N

  const dim3 gblk(32, 4);
  auto cdiv256 = [](size_t n) { return dim3((unsigned)((n + 255) / 256)); };

  // 0) one-shot bf16 conversions (x + all GEMM weights), 4 elems/thread
  cvt_bf16<<<cdiv256(X_N / 4),    dim3(256), 0, stream>>>(x,     x_bf,    X_N / 4);
  cvt_bf16<<<cdiv256(WIN_N / 4),  dim3(256), 0, stream>>>(W_in,  Win_bf,  WIN_N / 4);
  cvt_bf16<<<cdiv256(WX_N / 4),   dim3(256), 0, stream>>>(W_x,   Wx_bf,   WX_N / 4);
  cvt_bf16<<<cdiv256(WOUT_N / 4), dim3(256), 0, stream>>>(W_out, Wout_bf, WOUT_N / 4);

  // 1) xi = x @ W_in^T   (M=32768, N=2048, K=1024)
  gemm_wmma_bf16<<<dim3(D_INNER / 64, BL / 256), gblk, 0, stream>>>(
      x_bf, Win_bf, xi, D_INNER, D_MODEL);

  // 2) causal depthwise conv -> bf16
  conv_dw<<<cdiv256(XI_N), dim3(256), 0, stream>>>(xi, conv_w, conv_b, xc_bf);

  // 3) ssm = xc @ W_x^T  (M=32768, N=320, K=2048)
  gemm_wmma_bf16<<<dim3(SSM_W / 64, BL / 256), gblk, 0, stream>>>(
      xc_bf, Wx_bf, ssm, SSM_W, D_INNER);

  // 4) delta / A_bar / B_bar
  ssm_prep<<<cdiv256(ST_N), dim3(256), 0, stream>>>(
      ssm, W_dt, b_dt, A_log, abar, bbar);

  // 5) selective scan
  scan_kernel<<<dim3(B_SZ), dim3(D_STATE), 0, stream>>>(abar, bbar, h);

  // 6) y = <C, h> * w_gate
  y_kernel<<<cdiv256(BL), dim3(256), 0, stream>>>(ssm, h, w_gate, yv);

  // 7) gated activation -> bf16 (into xc_bf, dead after GEMM2)
  gate_kernel<<<cdiv256(XI_N), dim3(256), 0, stream>>>(xi, yv, xc_bf);

  // 8) out = g @ W_out^T (M=32768, N=1024, K=2048)
  gemm_wmma_bf16<<<dim3(D_MODEL / 64, BL / 256), gblk, 0, stream>>>(
      xc_bf, Wout_bf, out, D_MODEL, D_INNER);
}
